// WaveNet_75737453297766
// MI455X (gfx1250) — matrix operand order repaired and checked by reading, not compiled, not run on hardware
//
#include <hip/hip_runtime.h>
#include <hip/hip_bf16.h>
#include <stdint.h>

typedef __attribute__((ext_vector_type(16))) __bf16 v16bf;
typedef __attribute__((ext_vector_type(8)))  float  v8f;

#define R_CH   256
#define S_CH   128
#define O_CH   64
#define NOUT   24
#define BATCH  16
#define TLEN   2048
#define NLAYER 32
#define XW     1024   // x window buffer width (cols); col c == t = TLEN-XW+c
#define W0     1021   // receptive field of final timestep
#define NTILE  32     // output columns per block

// ---------- helpers ----------
__device__ inline unsigned short f2bf(float f) {
    unsigned int u = __float_as_uint(f);
    unsigned int r = (u + 0x7FFFu + ((u >> 16) & 1u)) >> 16;
    return (unsigned short)r;
}

__device__ inline v16bf load_frag(const unsigned short* p) {
    union { uint4 u[2]; v16bf v; } f;
    const uint4* q = (const uint4*)p;
    f.u[0] = q[0];
    f.u[1] = q[1];
    return f.v;
}

#define WMMA_BF16(A, B, C) \
    __builtin_amdgcn_wmma_f32_16x16x32_bf16(false, (A), false, (B), (short)0, (C), false, false)

// ---------- weight prepack: fp32 -> bf16 in WMMA A-fragment order ----------
// A-fragment (16x32 bf16): lane L: m = mt*16 + (L&15), h = L>>4,
// element j: K = kt*32 + 16*h + j
#define NF_PACK (32*16*16*32*16)   // w_filter / w_gate packed elems (K=512)
#define NR_PACK (32*8*16*32*16)    // w_res packed elems (K=256)

__global__ __launch_bounds__(256) void prepack_kernel(
    const float* __restrict__ wf, const float* __restrict__ wg,
    const float* __restrict__ wr,
    unsigned short* __restrict__ wfP, unsigned short* __restrict__ wgP,
    unsigned short* __restrict__ wrP)
{
    int idx = blockIdx.x * 256 + threadIdx.x;
    if (idx < 2 * NF_PACK) {
        const float* src = (idx < NF_PACK) ? wf : wg;
        unsigned short* dst = (idx < NF_PACK) ? wfP : wgP;
        int o = (idx < NF_PACK) ? idx : idx - NF_PACK;
        int e = o;
        int j    = e & 15; e >>= 4;
        int lane = e & 31; e >>= 5;
        int mt   = e & 15; e >>= 4;
        int kt   = e & 15; e >>= 4;
        int i    = e;
        int m  = mt * 16 + (lane & 15);
        int h  = lane >> 4;
        int kk = j + 16 * h;
        int K  = kt * 32 + kk;          // 0..511
        int tap = K >> 8;               // K<256 -> tap0 (x[t-d]), else tap1 (x[t])
        int c   = K & 255;
        dst[o] = f2bf(src[(((size_t)i * 256 + m) * 256 + c) * 2 + tap]);
    } else {
        int o = idx - 2 * NF_PACK;      // < NR_PACK
        int e = o;
        int j    = e & 15; e >>= 4;
        int lane = e & 31; e >>= 5;
        int mt   = e & 15; e >>= 4;
        int kt   = e & 7;  e >>= 3;
        int i    = e;
        int m  = mt * 16 + (lane & 15);
        int h  = lane >> 4;
        int kk = j + 16 * h;
        int K  = kt * 32 + kk;          // 0..255
        wrP[o] = f2bf(wr[((size_t)i * 256 + m) * 256 + K]);
    }
}

// ---------- input causal conv (F=32 -> R=256, k=2, d=1) over last W0 cols ----------
__global__ __launch_bounds__(256) void input_conv_kernel(
    const float* __restrict__ X,     // (B, T, F=32)
    const float* __restrict__ w_in,  // (R, 32, 2)
    const float* __restrict__ b_in,  // (R)
    float* __restrict__ x0)          // (B, R, XW)
{
    int blk = blockIdx.x;
    int b   = blk / W0;
    int ci  = blk % W0;
    int col = (XW - W0) + ci;        // 3..1023
    int t   = (TLEN - XW) + col;     // absolute time
    int r   = threadIdx.x;
    const float* xa = X + ((size_t)b * TLEN + (t - 1)) * 32;
    const float* xb = X + ((size_t)b * TLEN + t) * 32;
    const float* w  = w_in + (size_t)r * 64;
    float acc = b_in[r];
#pragma unroll
    for (int f = 0; f < 32; ++f) {
        acc += w[f * 2 + 0] * xa[f];
        acc += w[f * 2 + 1] * xb[f];
    }
    x0[((size_t)b * R_CH + r) * XW + col] = acc;
}

__global__ __launch_bounds__(256) void zero_skip_kernel(float* skip) {
    int i = blockIdx.x * 256 + threadIdx.x;
    if (i < BATCH * S_CH) skip[i] = 0.f;
}

// ---------- main dilated-gated layer ----------
// One block per (batch, 32-column tile counted from right edge).
// 8 waves; wave w computes output rows [32w, 32w+32) x 32 columns.
__global__ __launch_bounds__(256) void layer_kernel(
    const float* __restrict__ x_in,   // (B, R, XW) fp32
    float* __restrict__ x_out,        // (B, R, XW) fp32
    const unsigned short* __restrict__ wfL,  // layer's packed filter (16kt,16mt,32,16)
    const unsigned short* __restrict__ wgL,  // layer's packed gate
    const unsigned short* __restrict__ wrL,  // layer's packed res (8kt,16mt,32,16)
    const float* __restrict__ bF, const float* __restrict__ bG,
    const float* __restrict__ bR,
    float* __restrict__ fx_last,      // (B, R)
    int d, int colMin, int numTiles)
{
    __shared__ __align__(16) unsigned short Bs[512 * NTILE]; // 32 KB tap-pair B frags
    __shared__ __align__(16) unsigned short Zs[256 * NTILE]; // 16 KB f*g B frags

    const int tid  = threadIdx.x;
    const int b    = blockIdx.x / numTiles;
    const int tile = blockIdx.x % numTiles;
    const int colBase = XW - NTILE * (tile + 1);   // rightmost tile first

    // ---- stage B matrix (K=512 x N=32) into fragment layout, bf16 ----
    for (int e = 0; e < 64; ++e) {
        int idx = e * 256 + tid;        // 0..16383
        int k = idx >> 5;               // K row 0..511
        int n = idx & 31;               // column within tile
        int col = colBase + n;
        float v = 0.f;
        if (col >= colMin) {
            int c   = k & 255;
            int src = (k < 256) ? (col - d) : col;   // tap0 = x[t-d], tap1 = x[t]
            v = x_in[((size_t)b * R_CH + c) * XW + src];
        }
        int nt = n >> 4, n16 = n & 15;
        int kt = k >> 5, hh = (k >> 4) & 1, j = k & 15;
        Bs[(((nt * 16) + kt) * 32 + hh * 16 + n16) * 16 + j] = f2bf(v);
    }
    __syncthreads();

    const int wave = tid >> 5;
    const int lane = tid & 31;
    const int n16 = lane & 15;
    const int h = lane >> 4;
    const int mt0 = wave * 2;

    const v8f vzero = {0.f, 0.f, 0.f, 0.f, 0.f, 0.f, 0.f, 0.f};
    v8f accF[2][2] = {{vzero, vzero}, {vzero, vzero}};
    v8f accG[2][2] = {{vzero, vzero}, {vzero, vzero}};

    // ---- filter + gate GEMM: K=512, 16 k-steps, 8 WMMA per step ----
    // software pipeline: prefetch next k-step's A fragments before consuming cur
    size_t base0 = ((size_t)mt0 * 32 + lane) * 16;
    v16bf af[2], ag[2];
    af[0] = load_frag(wfL + base0);
    af[1] = load_frag(wfL + base0 + 512);
    ag[0] = load_frag(wgL + base0);
    ag[1] = load_frag(wgL + base0 + 512);
    v16bf nf[2] = {af[0], af[1]}, ng[2] = {ag[0], ag[1]};

    for (int kt = 0; kt < 16; ++kt) {
        v16bf b0 = load_frag(&Bs[((0 * 16 + kt) * 32 + lane) * 16]);
        v16bf b1 = load_frag(&Bs[((1 * 16 + kt) * 32 + lane) * 16]);
        if (kt < 15) {
            size_t nb = ((size_t)(kt + 1) * 16 + mt0) * 32 * 16 + (size_t)lane * 16;
            nf[0] = load_frag(wfL + nb);
            nf[1] = load_frag(wfL + nb + 512);
            ng[0] = load_frag(wgL + nb);
            ng[1] = load_frag(wgL + nb + 512);
        }
        accF[0][0] = WMMA_BF16(af[0], b0, accF[0][0]);
        accF[0][1] = WMMA_BF16(af[0], b1, accF[0][1]);
        accF[1][0] = WMMA_BF16(af[1], b0, accF[1][0]);
        accF[1][1] = WMMA_BF16(af[1], b1, accF[1][1]);
        accG[0][0] = WMMA_BF16(ag[0], b0, accG[0][0]);
        accG[0][1] = WMMA_BF16(ag[0], b1, accG[0][1]);
        accG[1][0] = WMMA_BF16(ag[1], b0, accG[1][0]);
        accG[1][1] = WMMA_BF16(ag[1], b1, accG[1][1]);
        af[0] = nf[0]; af[1] = nf[1];
        ag[0] = ng[0]; ag[1] = ng[1];
    }

    // ---- z = tanh(f) * sigmoid(g), restage as B fragments (K=256 x N=32) ----
    for (int mtI = 0; mtI < 2; ++mtI) {
        int mt = mt0 + mtI;
        for (int ntI = 0; ntI < 2; ++ntI) {
            for (int v = 0; v < 8; ++v) {
                int row = mt * 16 + v + 8 * h;
                float fv = tanhf(accF[mtI][ntI][v] + bF[row]);
                float gv = 1.f / (1.f + expf(-(accG[mtI][ntI][v] + bG[row])));
                float z  = fv * gv;
                int ktz = row >> 5;
                int h2  = (row >> 4) & 1;
                int j2  = row & 15;
                Zs[(((ntI * 8) + ktz) * 32 + h2 * 16 + n16) * 16 + j2] = f2bf(z);
            }
        }
    }
    __syncthreads();

    // ---- residual GEMM: K=256, 8 k-steps, 4 WMMA per step (pipelined) ----
    v8f accR[2][2] = {{vzero, vzero}, {vzero, vzero}};
    v16bf ar[2], nr[2];
    ar[0] = load_frag(wrL + base0);
    ar[1] = load_frag(wrL + base0 + 512);
    nr[0] = ar[0]; nr[1] = ar[1];
    for (int kt = 0; kt < 8; ++kt) {
        v16bf b0 = load_frag(&Zs[((0 * 8 + kt) * 32 + lane) * 16]);
        v16bf b1 = load_frag(&Zs[((1 * 8 + kt) * 32 + lane) * 16]);
        if (kt < 7) {
            size_t nb = ((size_t)(kt + 1) * 16 + mt0) * 32 * 16 + (size_t)lane * 16;
            nr[0] = load_frag(wrL + nb);
            nr[1] = load_frag(wrL + nb + 512);
        }
        accR[0][0] = WMMA_BF16(ar[0], b0, accR[0][0]);
        accR[0][1] = WMMA_BF16(ar[0], b1, accR[0][1]);
        accR[1][0] = WMMA_BF16(ar[1], b0, accR[1][0]);
        accR[1][1] = WMMA_BF16(ar[1], b1, accR[1][1]);
        ar[0] = nr[0]; ar[1] = nr[1];
    }

    // ---- fx + bias, residual add (fp32), store; capture fx at t = T-1 ----
    for (int ntI = 0; ntI < 2; ++ntI) {
        int col = colBase + ntI * 16 + n16;
        if (col < colMin) continue;
        for (int mtI = 0; mtI < 2; ++mtI) {
            int mt = mt0 + mtI;
            for (int v = 0; v < 8; ++v) {
                int row = mt * 16 + v + 8 * h;
                float fx = accR[mtI][ntI][v] + bR[row];
                size_t xi = ((size_t)b * R_CH + row) * XW + col;
                x_out[xi] = fx + x_in[xi];
                if (col == XW - 1) fx_last[b * R_CH + row] = fx;
            }
        }
    }
}

// ---------- skip accumulation at t = T-1 ----------
__global__ __launch_bounds__(128) void skip_kernel(
    const float* __restrict__ wsk,   // (S, R) for this layer
    const float* __restrict__ bsk,   // (S)
    const float* __restrict__ fx_last, // (B, R)
    float* __restrict__ skip)        // (B, S)
{
    int b = blockIdx.x;
    int s = threadIdx.x;
    const float* fx = fx_last + (size_t)b * R_CH;
    const float* w  = wsk + (size_t)s * R_CH;
    float acc = bsk[s];
#pragma unroll 4
    for (int r = 0; r < R_CH; ++r) acc += w[r] * fx[r];
    skip[b * S_CH + s] += acc;
}

// ---------- output head ----------
__global__ __launch_bounds__(128) void head_kernel(
    const float* __restrict__ skip,   // (B, S)
    const float* __restrict__ Xex,    // (B, 16)
    const float* __restrict__ w1, const float* __restrict__ b1,  // (O,S),(O)
    const float* __restrict__ w2, const float* __restrict__ b2,  // (O,O),(O)
    const float* __restrict__ wl, const float* __restrict__ bl,  // (NOUT,80),(NOUT)
    float* __restrict__ out)          // (B, NOUT)
{
    __shared__ float s1[S_CH];
    __shared__ float o1[O_CH];
    __shared__ float o2[O_CH];
    int b = blockIdx.x;
    int t = threadIdx.x;
    if (t < S_CH) s1[t] = fmaxf(skip[b * S_CH + t], 0.f);
    __syncthreads();
    if (t < O_CH) {
        float acc = b1[t];
        for (int s = 0; s < S_CH; ++s) acc += w1[t * S_CH + s] * s1[s];
        o1[t] = fmaxf(acc, 0.f);
    }
    __syncthreads();
    if (t < O_CH) {
        float acc = b2[t];
        for (int o = 0; o < O_CH; ++o) acc += w2[t * O_CH + o] * o1[o];
        o2[t] = acc;
    }
    __syncthreads();
    if (t < NOUT) {
        float acc = bl[t];
        for (int q = 0; q < O_CH; ++q) acc += wl[t * 80 + q] * fmaxf(o2[q], 0.f);
        for (int e = 0; e < 16; ++e)   acc += wl[t * 80 + 64 + e] * fmaxf(Xex[b * 16 + e], 0.f);
        out[b * NOUT + t] = acc;
    }
}

// ---------- host ----------
extern "C" void kernel_launch(void* const* d_in, const int* in_sizes, int n_in,
                              void* d_out, int out_size, void* d_ws, size_t ws_size,
                              hipStream_t stream)
{
    const float* X    = (const float*)d_in[0];
    const float* Xex  = (const float*)d_in[1];
    const float* w_in = (const float*)d_in[2];
    const float* b_in = (const float*)d_in[3];
    const float* wF   = (const float*)d_in[4];
    const float* bF   = (const float*)d_in[5];
    const float* wG   = (const float*)d_in[6];
    const float* bG   = (const float*)d_in[7];
    const float* wR   = (const float*)d_in[8];
    const float* bR   = (const float*)d_in[9];
    const float* wSk  = (const float*)d_in[10];
    const float* bSk  = (const float*)d_in[11];
    const float* w1   = (const float*)d_in[12];
    const float* b1   = (const float*)d_in[13];
    const float* w2   = (const float*)d_in[14];
    const float* b2   = (const float*)d_in[15];
    const float* wl   = (const float*)d_in[16];
    const float* bl   = (const float*)d_in[17];

    char* ws = (char*)d_ws;
    unsigned short* wfP = (unsigned short*)(ws);                       //  8 MiB
    unsigned short* wgP = (unsigned short*)(ws + 8388608);             //  8 MiB
    unsigned short* wrP = (unsigned short*)(ws + 16777216);            //  4 MiB
    float* xA   = (float*)(ws + 20971520);                             // 16 MiB
    float* xB   = (float*)(ws + 37748736);                             // 16 MiB
    float* fxL  = (float*)(ws + 54525952);                             // 16 KiB
    float* skip = (float*)(ws + 54542336);                             //  8 KiB

    // 1. pack weights to bf16 WMMA fragment layout
    prepack_kernel<<<(2 * NF_PACK + NR_PACK) / 256, 256, 0, stream>>>(
        wF, wG, wR, wfP, wgP, wrP);

    // 2. input conv over the 1021-column receptive window
    input_conv_kernel<<<BATCH * W0, 256, 0, stream>>>(X, w_in, b_in, xA);

    // 3. zero skip accumulator
    zero_skip_kernel<<<(BATCH * S_CH + 255) / 256, 256, 0, stream>>>(skip);

    // 4. 32 layers over shrinking windows, ping-pong x buffers
    float* xin = xA;
    float* xout = xB;
    int W = W0;
    for (int i = 0; i < NLAYER; ++i) {
        int d = 1 << (i % 8);
        int Wout = W - d;
        int nt = (Wout + NTILE - 1) / NTILE;
        layer_kernel<<<BATCH * nt, 256, 0, stream>>>(
            xin, xout,
            wfP + (size_t)i * 131072,
            wgP + (size_t)i * 131072,
            wrP + (size_t)i * 65536,
            bF + (size_t)i * R_CH, bG + (size_t)i * R_CH, bR + (size_t)i * R_CH,
            fxL, d, XW - Wout, nt);
        skip_kernel<<<BATCH, 128, 0, stream>>>(
            wSk + (size_t)i * S_CH * R_CH, bSk + (size_t)i * S_CH, fxL, skip);
        float* tmp = xin; xin = xout; xout = tmp;
        W = Wout;
    }

    // 5. head
    head_kernel<<<BATCH, 128, 0, stream>>>(skip, Xex, w1, b1, w2, b2, wl, bl,
                                           (float*)d_out);
    (void)in_sizes; (void)n_in; (void)out_size; (void)ws_size;
}
